// GRN_22299470201519
// MI455X (gfx1250) — matrix operand order repaired and checked by reading, not compiled, not run on hardware
//
#include <hip/hip_runtime.h>
#include <hip/hip_bf16.h>
#include <stddef.h>

// ---------------- WMMA types & helpers (gfx1250, wave32) ----------------
typedef __attribute__((ext_vector_type(16))) __bf16          v16bf;
typedef __attribute__((ext_vector_type(8)))  float           v8f;
typedef __attribute__((ext_vector_type(8)))  unsigned short  v8us;
typedef __attribute__((ext_vector_type(16))) unsigned short  v16us;

union FragU { v16bf bf; v16us us; v8us h2[2]; };

__device__ __forceinline__ unsigned short f2bf(float f) {
    unsigned u = __float_as_uint(f);
    unsigned r = (u + 0x7FFFu + ((u >> 16) & 1u)) >> 16;
    return (unsigned short)r;
}

__device__ __forceinline__ v8f zero8() {
    v8f z;
#pragma unroll
    for (int i = 0; i < 8; ++i) z[i] = 0.0f;
    return z;
}

// A-matrix 16x32 bf16 fragment: row m = lane&15; K map per ISA table:
// slots 0..7 -> K = 8*(lane>=16) + j ; slots 8..15 -> K = 16 + 8*(lane>=16) + (j-8)
__device__ __forceinline__ v16bf load_fragA(const unsigned short* base, int stride) {
    int lane = threadIdx.x & 31;
    const unsigned short* p = base + (size_t)(lane & 15) * stride + ((lane >> 4) << 3);
    FragU u;
    u.h2[0] = *(const v8us*)(p);
    u.h2[1] = *(const v8us*)(p + 16);
    return u.bf;
}

// B-matrix 32x16 bf16 fragment, loaded from B^T (rows = output col n, K contiguous):
// lane holds col n = lane&15; slots 0..7 -> K = 8*(lane>=16) + j ; slots 8..15 -> K = 16 + 8*(lane>=16) + (j-8)
__device__ __forceinline__ v16bf load_fragB(const unsigned short* baseT, int stride) {
    int lane = threadIdx.x & 31;
    const unsigned short* p = baseT + (size_t)(lane & 15) * stride + ((lane >> 4) << 3);
    FragU u;
    u.h2[0] = *(const v8us*)(p);
    u.h2[1] = *(const v8us*)(p + 16);
    return u.bf;
}

__device__ __forceinline__ v8f wmma_bf16(v16bf a, v16bf b, v8f c) {
    // D = A(16x32) * B(32x16) + C ; f32 accumulate
    return __builtin_amdgcn_wmma_f32_16x16x32_bf16(false, a, false, b, (short)0, c, false, false);
}

// ---------------- problem constants ----------------
#define NN   256
#define SS   32
#define MM   32
#define EE   32
#define BB   64
#define NIN  16
#define NOUT 16
#define ROWS (BB * NN)        // 16384
#define KBIG (NN * EE)        // 8192

// ---------------- prep kernels ----------------

// h[b,n,s]: inputs for n<16 else initial_state ; write f32 and bf16 copies
__global__ void k_build_h(const float* __restrict__ inputs, const float* __restrict__ init,
                          float* __restrict__ h32, unsigned short* __restrict__ hB) {
    int t = blockIdx.x * blockDim.x + threadIdx.x;           // < 524288
    int s = t & 31;
    int n = (t >> 5) & 255;
    float v;
    if (n < NIN) {
        int b = t >> 13;
        v = inputs[((size_t)b * NIN + n) * SS + s];
    } else {
        v = init[n * SS + s];
    }
    h32[t] = v;
    hB[t]  = f2bf(v);
}

// WfB[c][s] = bf16(Wf flat) ; (c = e*32+m, flat index identical)
__global__ void k_wfb(const float* __restrict__ Wf, unsigned short* __restrict__ wfB) {
    int t = blockIdx.x * blockDim.x + threadIdx.x;           // < 32768
    wfB[t] = f2bf(Wf[t]);
}

// edgeT[j][i*32+e] = bf16(edge[i][j][e])
__global__ void k_edgeT(const float* __restrict__ edge, unsigned short* __restrict__ edgeT) {
    int t = blockIdx.x * blockDim.x + threadIdx.x;           // < 2097152
    int j = t >> 13;
    int i = (t >> 5) & 255;
    int e = t & 31;
    edgeT[t] = f2bf(edge[((size_t)i << 13) + (j << 5) + e]);
}

// Transpose 6 GRU weight mats (32x32 f32) into bf16 WT[mat][n][k] = W[k][n]
__global__ void k_wtrans(const float* __restrict__ Wz, const float* __restrict__ Uz,
                         const float* __restrict__ Wr, const float* __restrict__ Ur,
                         const float* __restrict__ Wh, const float* __restrict__ Uh,
                         unsigned short* __restrict__ WT) {
    int t = blockIdx.x * blockDim.x + threadIdx.x;           // < 6144
    if (t >= 6 * 1024) return;
    int mi = t >> 10, idx = t & 1023;
    int n = idx >> 5, k = idx & 31;
    const float* src = (mi == 0) ? Wz : (mi == 1) ? Uz : (mi == 2) ? Wr
                     : (mi == 3) ? Ur : (mi == 4) ? Wh : Uh;
    WT[t] = f2bf(src[k * 32 + n]);
}

// Hs[b][s] = sum_i h[b,i,s]   (for the +bf rank-1 bias term)
__global__ void k_hs(const float* __restrict__ h32, float* __restrict__ Hs) {
    int t = blockIdx.x * blockDim.x + threadIdx.x;           // < 2048
    if (t >= BB * SS) return;
    int b = t >> 5, s = t & 31;
    float acc = 0.0f;
    for (int i = 0; i < NN; ++i) acc += h32[((size_t)b * NN + i) * SS + s];
    Hs[t] = acc;
}

// ---------------- GEMM 1: G[b,i,e,m] = sum_s h[b,i,s]*Wf[e,m*S+s] ----------------
// rows R=(b,i)=16384, cols c=(e,m)=1024, K=32 -> one wmma per 16x16 tile.
// Store transposed as GT[b][m][i*32+e] (bf16) so GEMM 2's B loads are contiguous.
__global__ void k_gfilt(const unsigned short* __restrict__ hB,
                        const unsigned short* __restrict__ wfB,
                        unsigned short* __restrict__ GT) {
    int wave = (blockIdx.x * blockDim.x + threadIdx.x) >> 5;   // 0..1023 row tiles
    int lane = threadIdx.x & 31;
    int col0 = lane & 15, rb = (lane >> 4) << 3;
    v16bf a = load_fragA(hB + (size_t)wave * 16 * SS, SS);
#pragma unroll 4
    for (int ct = 0; ct < 64; ++ct) {
        v16bf bfr = load_fragB(wfB + ct * 16 * SS, SS);
        v8f acc = zero8();
        acc = wmma_bf16(a, bfr, acc);
#pragma unroll
        for (int g = 0; g < 8; ++g) {
            int R = wave * 16 + rb + g;                        // (b,i)
            int c = ct * 16 + col0;                            // (e,m)
            int bI = R >> 8, i = R & 255;
            int e = c >> 5, m = c & 31;
            GT[(((size_t)bI * 32 + m) << 13) + i * 32 + e] = f2bf(acc[g]);
        }
    }
}

// ---------------- GEMM 2: messages[b,j,m] = sum_{(i,e)} edgeT[j][(i,e)] * GT[b][m][(i,e)] ----
// One wave per (b, j-tile): 16 rows x 32 cols, K=8192 -> 256 iters x 2 wmma.
__global__ void k_msg(const unsigned short* __restrict__ edgeT,
                      const unsigned short* __restrict__ GT,
                      const float* __restrict__ Hs, const float* __restrict__ bf,
                      unsigned short* __restrict__ msgB) {
    int wid = (blockIdx.x * blockDim.x + threadIdx.x) >> 5;    // 0..1023
    int b = wid >> 4;
    int jt = wid & 15;
    int lane = threadIdx.x & 31;
    int col0 = lane & 15, rb = (lane >> 4) << 3;

    const unsigned short* Abase = edgeT + (size_t)jt * 16 * KBIG;
    const unsigned short* B0 = GT + (((size_t)b * 32 + 0)  << 13);
    const unsigned short* B1 = GT + (((size_t)b * 32 + 16) << 13);

    v8f acc0 = zero8(), acc1 = zero8();
#pragma unroll 4
    for (int kb = 0; kb < KBIG; kb += 32) {
        v16bf a  = load_fragA(Abase + kb, KBIG);
        v16bf f0 = load_fragB(B0 + kb, KBIG);
        v16bf f1 = load_fragB(B1 + kb, KBIG);
        acc0 = wmma_bf16(a, f0, acc0);
        acc1 = wmma_bf16(a, f1, acc1);
        if (kb + 128 < KBIG)
            __builtin_prefetch(Abase + (size_t)(lane & 15) * KBIG + kb + 128, 0, 1);
    }

    // +bf filter-bias term: T[b,m] = sum_s Hs[b,s]*bf[m*32+s]  (independent of j)
    float t0 = 0.0f, t1 = 0.0f;
    for (int s = 0; s < SS; ++s) {
        float hs = Hs[b * SS + s];
        t0 += hs * bf[col0 * SS + s];
        t1 += hs * bf[(16 + col0) * SS + s];
    }
#pragma unroll
    for (int g = 0; g < 8; ++g) {
        int j = jt * 16 + rb + g;
        size_t row = (size_t)(b * NN + j) * MM;
        msgB[row + col0]      = f2bf(acc0[g] + t0);
        msgB[row + 16 + col0] = f2bf(acc1[g] + t1);
    }
}

// ---------------- GRU: z, r, h~ via WMMA; r*h crosses D->A layout through LDS ----------------
__global__ void k_gru(const unsigned short* __restrict__ msgB,
                      const unsigned short* __restrict__ hB,
                      const float* __restrict__ h32,
                      const unsigned short* __restrict__ WT,  // Wz,Uz,Wr,Ur,Wh,Uh (each 1024, transposed bf16)
                      const float* __restrict__ bz, const float* __restrict__ br,
                      const float* __restrict__ bh,
                      float* __restrict__ out_outputs, float* __restrict__ out_hnew) {
    __shared__ __align__(16) unsigned short ldsrh[4][16 * SS];
    int wid  = (blockIdx.x * blockDim.x + threadIdx.x) >> 5;   // 0..1023 row tile
    int wiw  = threadIdx.x >> 5;
    int lane = threadIdx.x & 31;
    int col0 = lane & 15, rb = (lane >> 4) << 3;
    size_t rowbase = (size_t)wid * 16 * SS;

    v16bf aM = load_fragA(msgB + rowbase, SS);
    v16bf aH = load_fragA(hB  + rowbase, SS);

    const unsigned short* WzT = WT + 0 * 1024;
    const unsigned short* UzT = WT + 1 * 1024;
    const unsigned short* WrT = WT + 2 * 1024;
    const unsigned short* UrT = WT + 3 * 1024;
    const unsigned short* WhT = WT + 4 * 1024;
    const unsigned short* UhT = WT + 5 * 1024;

    v8f z[2], hf[2];
#pragma unroll
    for (int ct = 0; ct < 2; ++ct) {
        v8f az = zero8();
        az = wmma_bf16(aM, load_fragB(WzT + ct * 16 * SS, SS), az);
        az = wmma_bf16(aH, load_fragB(UzT + ct * 16 * SS, SS), az);
        v8f ar = zero8();
        ar = wmma_bf16(aM, load_fragB(WrT + ct * 16 * SS, SS), ar);
        ar = wmma_bf16(aH, load_fragB(UrT + ct * 16 * SS, SS), ar);

        float bzc = bz[ct * 16 + col0], brc = br[ct * 16 + col0];
#pragma unroll
        for (int g = 0; g < 8; ++g) {
            int R = wid * 16 + rb + g;
            float hv = h32[(size_t)R * SS + ct * 16 + col0];
            hf[ct][g] = hv;
            z[ct][g]  = 1.0f / (1.0f + __expf(-(az[g] + bzc)));
            float rv  = 1.0f / (1.0f + __expf(-(ar[g] + brc)));
            ldsrh[wiw][(rb + g) * SS + ct * 16 + col0] = f2bf(rv * hv);  // D-layout -> LDS
        }
    }
    __syncthreads();
    v16bf aRH = load_fragA(&ldsrh[wiw][0], SS);                          // read back in A-layout

#pragma unroll
    for (int ct = 0; ct < 2; ++ct) {
        v8f ah = zero8();
        ah = wmma_bf16(aM,  load_fragB(WhT + ct * 16 * SS, SS), ah);
        ah = wmma_bf16(aRH, load_fragB(UhT + ct * 16 * SS, SS), ah);
        float bhc = bh[ct * 16 + col0];
#pragma unroll
        for (int g = 0; g < 8; ++g) {
            float e2 = __expf(2.0f * (ah[g] + bhc));
            float th = (e2 - 1.0f) / (e2 + 1.0f);
            float hn = (1.0f - z[ct][g]) * hf[ct][g] + z[ct][g] * th;
            int R = wid * 16 + rb + g;
            int scol = ct * 16 + col0;
            out_hnew[(size_t)R * SS + scol] = hn;
            int n = R & 255;
            if (n >= NN - NOUT) {                                        // neurons 240..255, reversed
                int b = R >> 8;
                int o = (NN - 1) - n;
                out_outputs[((size_t)o * BB + b) * SS + scol] = hn;
            }
        }
    }
}

// ---------------- launch ----------------
extern "C" void kernel_launch(void* const* d_in, const int* in_sizes, int n_in,
                              void* d_out, int out_size, void* d_ws, size_t ws_size,
                              hipStream_t stream) {
    const float* inputs = (const float*)d_in[0];
    const float* init   = (const float*)d_in[1];
    const float* edge   = (const float*)d_in[2];
    const float* Wf     = (const float*)d_in[3];
    const float* bf     = (const float*)d_in[4];
    const float* Wz     = (const float*)d_in[5];
    const float* Uz     = (const float*)d_in[6];
    const float* bz     = (const float*)d_in[7];
    const float* Wr     = (const float*)d_in[8];
    const float* Ur     = (const float*)d_in[9];
    const float* br     = (const float*)d_in[10];
    const float* Wh     = (const float*)d_in[11];
    const float* Uh     = (const float*)d_in[12];
    const float* bh     = (const float*)d_in[13];

    float* out_outputs = (float*)d_out;                         // [16,64,32]
    float* out_hnew    = (float*)d_out + NOUT * BB * SS;        // [64,256,32]

    char* ws = (char*)d_ws;
    float*          h32   = (float*)(ws + 0);                   //  2 MB  f32 h
    unsigned short* hB    = (unsigned short*)(ws + 2097152);    //  1 MB  bf16 h
    unsigned short* wfB   = (unsigned short*)(ws + 3145728);    // 64 KB  bf16 Wf
    unsigned short* edgeT = (unsigned short*)(ws + 3211264);    //  4 MB  bf16 edge^T
    unsigned short* GT    = (unsigned short*)(ws + 7405568);    // 32 MB  bf16 G (transposed)
    unsigned short* msgB  = (unsigned short*)(ws + 40960000);   //  1 MB  bf16 messages
    unsigned short* WT    = (unsigned short*)(ws + 42008576);   // 12 KB  bf16 GRU weights^T
    float*          Hs    = (float*)(ws + 42020864);            //  8 KB  row-sum of h

    k_build_h<<<2048, 256, 0, stream>>>(inputs, init, h32, hB);
    k_wfb    <<<128,  256, 0, stream>>>(Wf, wfB);
    k_edgeT  <<<8192, 256, 0, stream>>>(edge, edgeT);
    k_wtrans <<<24,   256, 0, stream>>>(Wz, Uz, Wr, Ur, Wh, Uh, WT);
    k_hs     <<<8,    256, 0, stream>>>(h32, Hs);

    k_gfilt  <<<256, 128, 0, stream>>>(hB, wfB, GT);            // 1024 waves, 64 wmma each
    k_msg    <<<256, 128, 0, stream>>>(edgeT, GT, Hs, bf, msgB);// 1024 waves, 512 wmma each
    k_gru    <<<256, 128, 0, stream>>>(msgB, hB, h32, WT, bz, br, bh,
                                       out_outputs, out_hnew);  // 1024 waves, 12 wmma each
}
